// StreamingLightDFN_35716948034361
// MI455X (gfx1250) — matrix operand order repaired and checked by reading, not compiled, not run on hardware
//
#include <hip/hip_runtime.h>
#include <math.h>

// ---------------- model constants ----------------
#define B_SZ   1024
#define C_CH   64
#define NB_ERB 32
#define NB_DF  96
#define ORDER  5
#define EMB    256
#define FREQ   481

typedef __attribute__((ext_vector_type(16))) _Float16 v16h;
typedef __attribute__((ext_vector_type(8)))  _Float16 v8h;
typedef __attribute__((ext_vector_type(8)))  float    v8f;

// =====================================================================
// Generic WMMA GEMM:  OUT[m,n] = epi( sum_k A[m,k] * W[n,k] + bias[n] )
//   A: f32 [M,K] row-major, W: f32 [N,K] row-major (i.e. x @ W.T)
//   epi: act==1 -> relu;  res!=null -> out = res + act(gemm+bias)
// Requires M%64==0, N%64==0, K%32==0 (true for all call sites here).
// Block tile 64x64, 8 waves; each wave computes 16x32 (2 WMMAs per
// 32-K step, A-fragment reuse). LDS double-buffered: one barrier per
// K step, staging of tile k+1 overlaps WMMA of tile k; tile k+2 is
// prefetched (global_prefetch_b8).
// =====================================================================
#define TM 64
#define TN 64
#define TK 32

__global__ __launch_bounds__(256) void gemm_wmma_kernel(
    const float* __restrict__ A, const float* __restrict__ W,
    const float* __restrict__ bias, const float* __restrict__ res,
    float* __restrict__ out, int M, int N, int K, int act)
{
  __shared__ __attribute__((aligned(64))) _Float16 As[2][TM][TK]; // 8 KB
  __shared__ __attribute__((aligned(64))) _Float16 Bs[2][TN][TK]; // 8 KB

  const int tid  = threadIdx.x;
  const int lane = tid & 31;
  const int wave = tid >> 5;            // 0..7
  const int mw   = (wave >> 1) * 16;    // 0,16,32,48
  const int nw   = (wave & 1) * 32;     // 0,32
  const int bm   = blockIdx.y * TM;
  const int bn   = blockIdx.x * TN;

  // staging assignment: 64 rows x 32 cols, 8 f32->f16 converts per thread
  const int sr = tid >> 2;              // 0..63
  const int sc = (tid & 3) * 8;         // 0,8,16,24
  const float* aptr = A + (long)(bm + sr) * K + sc;
  const float* wptr = W + (long)(bn + sr) * K + sc;

  const int mrow = mw + (lane & 15);
  const int nrow0 = nw + (lane & 15);
  const int nrow1 = nw + 16 + (lane & 15);
  const int hi   = lane >> 4;

  v8f acc0 = {}, acc1 = {};
  const int nk = K / TK;

  // stage tile 0
  {
#pragma unroll
    for (int i = 0; i < 8; ++i) As[0][sr][sc + i] = (_Float16)aptr[i];
#pragma unroll
    for (int i = 0; i < 8; ++i) Bs[0][sr][sc + i] = (_Float16)wptr[i];
  }
  __syncthreads();

  for (int kt = 0; kt < nk; ++kt) {
    // stage tile kt+1 into the other buffer (overlaps WMMA below)
    if (kt + 1 < nk) {
      const int nb = (kt + 1) & 1;
      const float* sa = aptr + (kt + 1) * TK;
      const float* sw = wptr + (kt + 1) * TK;
#pragma unroll
      for (int i = 0; i < 8; ++i) As[nb][sr][sc + i] = (_Float16)sa[i];
#pragma unroll
      for (int i = 0; i < 8; ++i) Bs[nb][sr][sc + i] = (_Float16)sw[i];
    }
    // prefetch tile kt+2 toward the caches
    if (kt + 2 < nk) {
      __builtin_prefetch(aptr + (kt + 2) * TK, 0, 1);
      __builtin_prefetch(wptr + (kt + 2) * TK, 0, 1);
    }

    const int cb = kt & 1;
    // A fragment (16x32 f16): lane half hi=0 holds K 0..15,
    // hi=1 holds K 16..31 for its row (contiguous, matches B).
    v8h alo = *reinterpret_cast<const v8h*>(&As[cb][mrow][16 * hi]);
    v8h ahi = *reinterpret_cast<const v8h*>(&As[cb][mrow][16 * hi + 8]);
    v16h a = __builtin_shufflevector(alo, ahi,
              0,1,2,3,4,5,6,7,8,9,10,11,12,13,14,15);
    // B fragments (32x16 f16): lane n holds column n (= row of W tile),
    // K contiguous per lane half (hi=0 -> K0..15, hi=1 -> K16..31).
    v16h b0 = *reinterpret_cast<const v16h*>(&Bs[cb][nrow0][16 * hi]);
    v16h b1 = *reinterpret_cast<const v16h*>(&Bs[cb][nrow1][16 * hi]);

    acc0 = __builtin_amdgcn_wmma_f32_16x16x32_f16(
        false, a, false, b0, (short)0, acc0, false, false);
    acc1 = __builtin_amdgcn_wmma_f32_16x16x32_f16(
        false, a, false, b1, (short)0, acc1, false, false);
    __syncthreads();
  }

  // epilogue: C/D layout -> VGPR v holds row mw+v+8*hi, col = (lane&15)
  const int gn0 = bn + nw + (lane & 15);
  const int gn1 = gn0 + 16;
  const float bv0 = bias ? bias[gn0] : 0.0f;
  const float bv1 = bias ? bias[gn1] : 0.0f;
#pragma unroll
  for (int v = 0; v < 8; ++v) {
    int gm = bm + mw + v + 8 * hi;
    float v0 = acc0[v] + bv0;
    float v1 = acc1[v] + bv1;
    if (act == 1) { v0 = fmaxf(v0, 0.0f); v1 = fmaxf(v1, 0.0f); }
    if (res) {
      v0 += res[(long)gm * N + gn0];
      v1 += res[(long)gm * N + gn1];
    }
    out[(long)gm * N + gn0] = v0;
    out[(long)gm * N + gn1] = v1;
  }
}

// =====================================================================
// GRU pointwise: hnew = (1-z)*n + z*h from precomputed gate GEMMs
// =====================================================================
__global__ void gru_pointwise_kernel(
    const float* __restrict__ gi, const float* __restrict__ gh,
    const float* __restrict__ h, float* __restrict__ hnew, int H, int total)
{
  int i = blockIdx.x * blockDim.x + threadIdx.x;
  if (i >= total) return;
  int b = i / H, j = i - b * H;
  const float* gib = gi + (long)b * 3 * H;
  const float* ghb = gh + (long)b * 3 * H;
  float ir = gib[j], iz = gib[H + j], in = gib[2 * H + j];
  float hr = ghb[j], hz = ghb[H + j], hn = ghb[2 * H + j];
  float r = 1.0f / (1.0f + expf(-(ir + hr)));
  float z = 1.0f / (1.0f + expf(-(iz + hz)));
  float n = tanhf(in + r * hn);
  hnew[(long)b * H + j] = (1.0f - z) * n + z * h[(long)b * H + j];
}

// =====================================================================
// Encoder: one block per batch item. ERB conv chain + DF conv chain,
// all intermediates LDS-resident. Emits e0..e3, c0, c1-flat, emb0 and
// the rotated buf_erb0 / buf_df0.
// =====================================================================
__global__ __launch_bounds__(256) void encoder_kernel(
    const float* __restrict__ feat_erb, const float* __restrict__ buf_erb0,
    const float* __restrict__ feat_spec, const float* __restrict__ buf_df0,
    const float* __restrict__ w_e0, const float* __restrict__ b_e0,
    const float* __restrict__ w_e1, const float* __restrict__ b_e1,
    const float* __restrict__ w_e2, const float* __restrict__ b_e2,
    const float* __restrict__ w_e3, const float* __restrict__ b_e3,
    const float* __restrict__ w_d0, const float* __restrict__ b_d0,
    const float* __restrict__ w_d1, const float* __restrict__ b_d1,
    float* __restrict__ e0g, float* __restrict__ e1g,
    float* __restrict__ e2g, float* __restrict__ e3g,
    float* __restrict__ emb0, float* __restrict__ c0g,
    float* __restrict__ c1flat,
    float* __restrict__ out_buf_erb0, float* __restrict__ out_buf_df0)
{
  const int b = blockIdx.x;
  const int tid = threadIdx.x;
  __shared__ float ctx_e[3][NB_ERB];
  __shared__ float ctx_d[2][3][NB_DF];
  __shared__ float se0[C_CH][NB_ERB];
  __shared__ float se1[C_CH][16];
  __shared__ float se2[C_CH][8];
  __shared__ float sc0[C_CH][NB_DF];

  // context assembly (time concat) + streaming-buffer rotation
  if (tid < 96) {
    int t = tid >> 5, f = tid & 31;
    ctx_e[t][f] = (t < 2) ? buf_erb0[b * 64 + t * 32 + f] : feat_erb[b * 32 + f];
  }
  if (tid < 64) {
    int t = tid >> 5, f = tid & 31;
    out_buf_erb0[b * 64 + tid] =
        (t == 0) ? buf_erb0[b * 64 + 32 + f] : feat_erb[b * 32 + f];
  }
  for (int idx = tid; idx < 2 * 3 * NB_DF; idx += 256) {
    int c = idx / (3 * NB_DF);
    int r = idx - c * 3 * NB_DF;
    int t = r / NB_DF, f = r - t * NB_DF;
    ctx_d[c][t][f] = (t < 2) ? buf_df0[b * 384 + c * 192 + t * 96 + f]
                             : feat_spec[b * 192 + f * 2 + c];
  }
  for (int idx = tid; idx < 384; idx += 256) {
    int c = idx / 192;
    int r = idx - c * 192;
    int t = r / 96, f = r - t * 96;
    out_buf_df0[b * 384 + idx] = (t == 0)
        ? buf_df0[b * 384 + c * 192 + 96 + f]
        : feat_spec[b * 192 + f * 2 + c];
  }
  __syncthreads();

  // e0 = relu(conv3x3(ctx_e)), pad freq 1, valid time
  for (int idx = tid; idx < C_CH * NB_ERB; idx += 256) {
    int c = idx >> 5, f = idx & 31;
    float acc = b_e0[c];
#pragma unroll
    for (int kt = 0; kt < 3; ++kt)
#pragma unroll
      for (int kf = 0; kf < 3; ++kf) {
        int ff = f + kf - 1;
        if (ff >= 0 && ff < NB_ERB)
          acc += ctx_e[kt][ff] * w_e0[c * 9 + kt * 3 + kf];
      }
    float v = fmaxf(acc, 0.f);
    se0[c][f] = v;
    e0g[((long)b * C_CH + c) * NB_ERB + f] = v;
  }
  // c0 = relu(conv3x3(ctx_d)), 2 in-channels
  for (int idx = tid; idx < C_CH * NB_DF; idx += 256) {
    int c = idx / NB_DF, f = idx - c * NB_DF;
    float acc = b_d0[c];
#pragma unroll
    for (int ci = 0; ci < 2; ++ci)
#pragma unroll
      for (int kt = 0; kt < 3; ++kt)
#pragma unroll
        for (int kf = 0; kf < 3; ++kf) {
          int ff = f + kf - 1;
          if (ff >= 0 && ff < NB_DF)
            acc += ctx_d[ci][kt][ff] * w_d0[c * 18 + ci * 9 + kt * 3 + kf];
        }
    float v = fmaxf(acc, 0.f);
    sc0[c][f] = v;
    c0g[((long)b * C_CH + c) * NB_DF + f] = v;
  }
  __syncthreads();

  // e1 = relu(1x2 stride-2 conv over freq)
  for (int idx = tid; idx < C_CH * 16; idx += 256) {
    int c = idx >> 4, f = idx & 15;
    float acc = b_e1[c];
    for (int ci = 0; ci < C_CH; ++ci)
      acc += se0[ci][2 * f]     * w_e1[(c * C_CH + ci) * 2]
           + se0[ci][2 * f + 1] * w_e1[(c * C_CH + ci) * 2 + 1];
    float v = fmaxf(acc, 0.f);
    se1[c][f] = v;
    e1g[((long)b * C_CH + c) * 16 + f] = v;
  }
  // c1 -> flattened (freq-major) A-matrix for df_fc_emb GEMM
  for (int idx = tid; idx < C_CH * 48; idx += 256) {
    int c = idx / 48, f = idx - c * 48;
    float acc = b_d1[c];
    for (int ci = 0; ci < C_CH; ++ci)
      acc += sc0[ci][2 * f]     * w_d1[(c * C_CH + ci) * 2]
           + sc0[ci][2 * f + 1] * w_d1[(c * C_CH + ci) * 2 + 1];
    c1flat[(long)b * 3072 + f * C_CH + c] = fmaxf(acc, 0.f);
  }
  __syncthreads();

  for (int idx = tid; idx < C_CH * 8; idx += 256) {
    int c = idx >> 3, f = idx & 7;
    float acc = b_e2[c];
    for (int ci = 0; ci < C_CH; ++ci)
      acc += se1[ci][2 * f]     * w_e2[(c * C_CH + ci) * 2]
           + se1[ci][2 * f + 1] * w_e2[(c * C_CH + ci) * 2 + 1];
    float v = fmaxf(acc, 0.f);
    se2[c][f] = v;
    e2g[((long)b * C_CH + c) * 8 + f] = v;
  }
  __syncthreads();

  for (int idx = tid; idx < C_CH * 4; idx += 256) {
    int c = idx >> 2, f = idx & 3;
    float acc = b_e3[c];
    for (int ci = 0; ci < C_CH; ++ci)
      acc += se2[ci][2 * f]     * w_e3[(c * C_CH + ci) * 2]
           + se2[ci][2 * f + 1] * w_e3[(c * C_CH + ci) * 2 + 1];
    float v = fmaxf(acc, 0.f);
    e3g[((long)b * C_CH + c) * 4 + f] = v;
    emb0[(long)b * EMB + f * C_CH + c] = v;   // transpose(0,2,3,1) flatten
  }
}

// =====================================================================
// Decoder: one block per batch item. 1x1 convs + freq-2x transposed
// convs (upsample: out[2i+s] = sum_ci x[ci][i] * w[co,ci,0,s]) -> mask.
// =====================================================================
__global__ __launch_bounds__(256) void decoder_kernel(
    const float* __restrict__ e0g, const float* __restrict__ e1g,
    const float* __restrict__ e2g, const float* __restrict__ e3g,
    const float* __restrict__ h_erb,
    const float* __restrict__ w3p, const float* __restrict__ b3p,
    const float* __restrict__ wt3, const float* __restrict__ bt3,
    const float* __restrict__ w2p, const float* __restrict__ b2p,
    const float* __restrict__ wt2, const float* __restrict__ bt2,
    const float* __restrict__ w1p, const float* __restrict__ b1p,
    const float* __restrict__ wt1, const float* __restrict__ bt1,
    const float* __restrict__ w0p, const float* __restrict__ b0p,
    const float* __restrict__ w0o, const float* __restrict__ b0o,
    float* __restrict__ maskg)
{
  const int b = blockIdx.x;
  const int tid = threadIdx.x;
  __shared__ float se3[C_CH][4], se2[C_CH][8], se1[C_CH][16], se0[C_CH][32];
  __shared__ float x3[C_CH][4], d3[C_CH][8];
  __shared__ float x2[C_CH][8], d2[C_CH][16];
  __shared__ float x1[C_CH][16], d1[C_CH][32];
  __shared__ float x0[C_CH][32];

  for (int i = tid; i < C_CH * 4;  i += 256) se3[i >> 2][i & 3]  = e3g[(long)b * C_CH * 4 + i];
  for (int i = tid; i < C_CH * 8;  i += 256) se2[i >> 3][i & 7]  = e2g[(long)b * C_CH * 8 + i];
  for (int i = tid; i < C_CH * 16; i += 256) se1[i >> 4][i & 15] = e1g[(long)b * C_CH * 16 + i];
  for (int i = tid; i < C_CH * 32; i += 256) se0[i >> 5][i & 31] = e0g[(long)b * C_CH * 32 + i];
  __syncthreads();

  { // x3 = conv3p(e3) + emb_erb
    int c = tid >> 2, f = tid & 3;
    float acc = b3p[c] + h_erb[(long)b * EMB + f * C_CH + c];
    for (int ci = 0; ci < C_CH; ++ci) acc += se3[ci][f] * w3p[c * C_CH + ci];
    x3[c][f] = acc;
  }
  __syncthreads();
  for (int idx = tid; idx < 512; idx += 256) {
    int c = idx >> 3, of = idx & 7, f = of >> 1, s = of & 1;
    float acc = bt3[c];
    for (int ci = 0; ci < C_CH; ++ci) acc += x3[ci][f] * wt3[(c * C_CH + ci) * 2 + s];
    d3[c][of] = fmaxf(acc, 0.f);
  }
  __syncthreads();
  for (int idx = tid; idx < 512; idx += 256) {
    int c = idx >> 3, f = idx & 7;
    float acc = b2p[c] + d3[c][f];
    for (int ci = 0; ci < C_CH; ++ci) acc += se2[ci][f] * w2p[c * C_CH + ci];
    x2[c][f] = acc;
  }
  __syncthreads();
  for (int idx = tid; idx < 1024; idx += 256) {
    int c = idx >> 4, of = idx & 15, f = of >> 1, s = of & 1;
    float acc = bt2[c];
    for (int ci = 0; ci < C_CH; ++ci) acc += x2[ci][f] * wt2[(c * C_CH + ci) * 2 + s];
    d2[c][of] = fmaxf(acc, 0.f);
  }
  __syncthreads();
  for (int idx = tid; idx < 1024; idx += 256) {
    int c = idx >> 4, f = idx & 15;
    float acc = b1p[c] + d2[c][f];
    for (int ci = 0; ci < C_CH; ++ci) acc += se1[ci][f] * w1p[c * C_CH + ci];
    x1[c][f] = acc;
  }
  __syncthreads();
  for (int idx = tid; idx < 2048; idx += 256) {
    int c = idx >> 5, of = idx & 31, f = of >> 1, s = of & 1;
    float acc = bt1[c];
    for (int ci = 0; ci < C_CH; ++ci) acc += x1[ci][f] * wt1[(c * C_CH + ci) * 2 + s];
    d1[c][of] = fmaxf(acc, 0.f);
  }
  __syncthreads();
  for (int idx = tid; idx < 2048; idx += 256) {
    int c = idx >> 5, f = idx & 31;
    float acc = b0p[c] + d1[c][f];
    for (int ci = 0; ci < C_CH; ++ci) acc += se0[ci][f] * w0p[c * C_CH + ci];
    x0[c][f] = acc;
  }
  __syncthreads();

  if (tid < NB_ERB) {
    int f = tid;
    float acc = b0o[0];
    for (int c = 0; c < C_CH; ++c)
#pragma unroll
      for (int kf = 0; kf < 3; ++kf) {
        int ff = f + kf - 1;
        if (ff >= 0 && ff < NB_ERB) acc += x0[c][ff] * w0o[c * 3 + kf];
      }
    maskg[b * NB_ERB + f] = 1.f / (1.f + expf(-acc));
  }
}

// =====================================================================
// mask expansion (mask @ erb_inv), spec masking (f >= NB_DF) and
// buf_spec rotation.
// =====================================================================
__global__ void mask_apply_kernel(
    const float* __restrict__ maskg, const float* __restrict__ erb_inv,
    const float* __restrict__ spec, const float* __restrict__ buf_spec,
    float* __restrict__ spec_e, float* __restrict__ out_buf_spec)
{
  int idx = blockIdx.x * blockDim.x + threadIdx.x;
  if (idx >= B_SZ * FREQ) return;
  int b = idx / FREQ, f = idx - b * FREQ;
  float m = 0.f;
#pragma unroll
  for (int e = 0; e < NB_ERB; ++e) m += maskg[b * NB_ERB + e] * erb_inv[e * FREQ + f];
  long sb = (long)b * 2 * FREQ;
  float sr = spec[sb + f * 2], si = spec[sb + f * 2 + 1];
  if (f >= NB_DF) {
    spec_e[sb + f * 2]     = sr * m;
    spec_e[sb + f * 2 + 1] = si * m;
  }
  long base = (long)b * 4 * 2 * FREQ;
#pragma unroll
  for (int t = 0; t < 3; ++t) {
    out_buf_spec[base + t * 2 * FREQ + f * 2]     = buf_spec[base + (t + 1) * 2 * FREQ + f * 2];
    out_buf_spec[base + t * 2 * FREQ + f * 2 + 1] = buf_spec[base + (t + 1) * 2 * FREQ + f * 2 + 1];
  }
  out_buf_spec[base + 3 * 2 * FREQ + f * 2]     = sr;
  out_buf_spec[base + 3 * 2 * FREQ + f * 2 + 1] = si;
}

// =====================================================================
// df final: buf_dfp rotation, df_convp (5-tap time conv), tanh coefs,
// order-5 complex deep-filter, writes spec_e[f < NB_DF].
// =====================================================================
__global__ __launch_bounds__(256) void df_final_kernel(
    const float* __restrict__ c2,       // [B,960] df_out GEMM
    const float* __restrict__ c0g,      // [B,64,96]
    const float* __restrict__ buf_dfp,  // [B,64,4,96]
    const float* __restrict__ buf_spec, // [B,1,4,481,2]
    const float* __restrict__ spec,     // [B,1,1,481,2]
    const float* __restrict__ wp, const float* __restrict__ bp,
    float* __restrict__ spec_e, float* __restrict__ out_buf_dfp)
{
  const int b = blockIdx.x;
  const int tid = threadIdx.x;
  __shared__ float swp[10 * C_CH * ORDER];   // df_convp_w (10,64,5,1)
  for (int i = tid; i < 10 * C_CH * ORDER; i += 256) swp[i] = wp[i];

  // rotate buf_dfp (ctx_p[:, :, 1:]): t<3 from old buf, t==3 = c0
  for (int idx = tid; idx < C_CH * 4 * NB_DF; idx += 256) {
    int ci = idx / 384;
    int r  = idx - ci * 384;
    int t  = r / 96, f = r - t * 96;
    float v = (t < 3) ? buf_dfp[(long)b * 24576 + ci * 384 + (t + 1) * 96 + f]
                      : c0g[((long)b * C_CH + ci) * 96 + f];
    out_buf_dfp[(long)b * 24576 + idx] = v;
  }
  __syncthreads();

  if (tid < NB_DF) {
    const int f = tid;
    float acc[10];
#pragma unroll
    for (int j = 0; j < 10; ++j) acc[j] = bp[j];
    for (int ci = 0; ci < C_CH; ++ci) {
#pragma unroll
      for (int t = 0; t < ORDER; ++t) {
        float v = (t < 4) ? buf_dfp[(long)b * 24576 + ci * 384 + t * 96 + f]
                          : c0g[((long)b * C_CH + ci) * 96 + f];
#pragma unroll
        for (int j = 0; j < 10; ++j) acc[j] += v * swp[j * 320 + ci * 5 + t];
      }
    }
    float coef[10];
#pragma unroll
    for (int j = 0; j < 10; ++j)
      coef[j] = tanhf(c2[(long)b * 960 + f * 10 + j] + acc[j]);

    float outr = 0.f, outi = 0.f;
#pragma unroll
    for (int o = 0; o < ORDER; ++o) {
      float sr, si;
      if (o < 4) {
        sr = buf_spec[(long)b * 3848 + o * 962 + f * 2];
        si = buf_spec[(long)b * 3848 + o * 962 + f * 2 + 1];
      } else {
        sr = spec[(long)b * 962 + f * 2];
        si = spec[(long)b * 962 + f * 2 + 1];
      }
      float cr = coef[o * 2], cim = coef[o * 2 + 1];
      outr += sr * cr - si * cim;
      outi += sr * cim + si * cr;
    }
    spec_e[(long)b * 962 + f * 2]     = outr;
    spec_e[(long)b * 962 + f * 2 + 1] = outi;
  }
}

// =====================================================================
extern "C" void kernel_launch(void* const* d_in, const int* in_sizes, int n_in,
                              void* d_out, int out_size, void* d_ws, size_t ws_size,
                              hipStream_t stream)
{
  (void)in_sizes; (void)n_in; (void)out_size; (void)ws_size;

  const float* spec      = (const float*)d_in[0];
  const float* feat_erb  = (const float*)d_in[1];
  const float* feat_spec = (const float*)d_in[2];
  const float* h_enc     = (const float*)d_in[3];
  const float* h_erb     = (const float*)d_in[4];
  const float* h_df      = (const float*)d_in[5];
  const float* buf_erb0  = (const float*)d_in[6];
  const float* buf_df0   = (const float*)d_in[7];
  const float* buf_dfp   = (const float*)d_in[8];
  const float* buf_spec  = (const float*)d_in[9];
  auto P = [&](int i) { return (const float*)d_in[10 + i]; };
  // 0 erb_conv0_w .. 7 erb_conv3_b, 8 df_conv0_w..11 df_conv1_b,
  // 12 df_fc_emb_w,13 b, 14-17 enc_gru, 18-21 erb_gru, 22-25 df_gru,
  // 26 df_skip_w,27 b, 28-43 decoder convs, 44 df_out_w,45 b,
  // 46 df_convp_w,47 b, 48 erb_inv

  // ---- output offsets (flat, return order) ----
  float* out = (float*)d_out;
  const long o_spec_e = 0;                          // 1024*962
  const long o_henc   = (long)B_SZ * 962;
  const long o_herb   = o_henc + (long)B_SZ * EMB;
  const long o_hdf    = o_herb + (long)B_SZ * EMB;
  const long o_berb   = o_hdf  + (long)B_SZ * EMB;
  const long o_bdf0   = o_berb + (long)B_SZ * 64;
  const long o_bdfp   = o_bdf0 + (long)B_SZ * 384;
  const long o_bspec  = o_bdfp + (long)B_SZ * 24576;

  // ---- workspace layout (floats) ----
  float* ws = (float*)d_ws;
  long off = 0;
  float* s_e0     = ws + off; off += (long)B_SZ * C_CH * 32;
  float* s_e1     = ws + off; off += (long)B_SZ * C_CH * 16;
  float* s_e2     = ws + off; off += (long)B_SZ * C_CH * 8;
  float* s_e3     = ws + off; off += (long)B_SZ * C_CH * 4;
  float* s_c0     = ws + off; off += (long)B_SZ * C_CH * NB_DF;
  float* s_c1flat = ws + off; off += (long)B_SZ * 3072;
  float* s_emb0   = ws + off; off += (long)B_SZ * EMB;
  float* s_emb    = ws + off; off += (long)B_SZ * EMB;
  float* s_gi     = ws + off; off += (long)B_SZ * 3 * EMB;
  float* s_gh     = ws + off; off += (long)B_SZ * 3 * EMB;
  float* s_cskip  = ws + off; off += (long)B_SZ * EMB;
  float* s_c2     = ws + off; off += (long)B_SZ * 960;
  float* s_mask   = ws + off; off += (long)B_SZ * NB_ERB;

  const dim3 blk(256);

  auto gemm = [&](const float* A, const float* W, const float* bias,
                  const float* res, float* o, int M, int N, int K, int act) {
    dim3 grid(N / TN, M / TM);
    gemm_wmma_kernel<<<grid, blk, 0, stream>>>(A, W, bias, res, o, M, N, K, act);
  };

  // 1) encoder conv chains
  encoder_kernel<<<dim3(B_SZ), blk, 0, stream>>>(
      feat_erb, buf_erb0, feat_spec, buf_df0,
      P(0), P(1), P(2), P(3), P(4), P(5), P(6), P(7),
      P(8), P(9), P(10), P(11),
      s_e0, s_e1, s_e2, s_e3, s_emb0, s_c0, s_c1flat,
      out + o_berb, out + o_bdf0);

  // 2) emb = e3_flat + relu(c1_flat @ df_fc_emb_w.T + b)
  gemm(s_c1flat, P(12), P(13), s_emb0, s_emb, B_SZ, EMB, 3072, 1);

  // 3) enc GRU
  gemm(s_emb, P(14), P(16), nullptr, s_gi, B_SZ, 3 * EMB, EMB, 0);
  gemm(h_enc, P(15), P(17), nullptr, s_gh, B_SZ, 3 * EMB, EMB, 0);
  gru_pointwise_kernel<<<dim3((B_SZ * EMB) / 256), blk, 0, stream>>>(
      s_gi, s_gh, h_enc, out + o_henc, EMB, B_SZ * EMB);

  // 4) erb GRU (input = h_enc_new)
  gemm(out + o_henc, P(18), P(20), nullptr, s_gi, B_SZ, 3 * EMB, EMB, 0);
  gemm(h_erb,        P(19), P(21), nullptr, s_gh, B_SZ, 3 * EMB, EMB, 0);
  gru_pointwise_kernel<<<dim3((B_SZ * EMB) / 256), blk, 0, stream>>>(
      s_gi, s_gh, h_erb, out + o_herb, EMB, B_SZ * EMB);

  // 5) df GRU (input = h_enc_new)
  gemm(out + o_henc, P(22), P(24), nullptr, s_gi, B_SZ, 3 * EMB, EMB, 0);
  gemm(h_df,         P(23), P(25), nullptr, s_gh, B_SZ, 3 * EMB, EMB, 0);
  gru_pointwise_kernel<<<dim3((B_SZ * EMB) / 256), blk, 0, stream>>>(
      s_gi, s_gh, h_df, out + o_hdf, EMB, B_SZ * EMB);

  // 6) c = h_df_new + emb @ df_skip.T + b ; c2 = c @ df_out.T + b
  gemm(out + o_henc, P(26), P(27), out + o_hdf, s_cskip, B_SZ, EMB, EMB, 0);
  gemm(s_cskip, P(44), P(45), nullptr, s_c2, B_SZ, 960, EMB, 0);

  // 7) decoder -> mask
  decoder_kernel<<<dim3(B_SZ), blk, 0, stream>>>(
      s_e0, s_e1, s_e2, s_e3, out + o_herb,
      P(28), P(29), P(30), P(31), P(32), P(33), P(34), P(35),
      P(36), P(37), P(38), P(39), P(40), P(41), P(42), P(43),
      s_mask);

  // 8) mask expansion + spec masking + buf_spec rotation
  mask_apply_kernel<<<dim3((B_SZ * FREQ + 255) / 256), blk, 0, stream>>>(
      s_mask, P(48), spec, buf_spec, out + o_spec_e, out + o_bspec);

  // 9) deep-filter coefficients + complex filter + buf_dfp rotation
  df_final_kernel<<<dim3(B_SZ), blk, 0, stream>>>(
      s_c2, s_c0, buf_dfp, buf_spec, spec, P(46), P(47),
      out + o_spec_e, out + o_bdfp);
}
